// PraxisSMEAR_70901320122578
// MI455X (gfx1250) — matrix operand order repaired and checked by reading, not compiled, not run on hardware
//
#include <hip/hip_runtime.h>
#include <cstdint>

#define B_ 4
#define S_ 2048
#define H_ 1024
#define FF_ 4096
#define E_ 8
#define T_ (B_*S_)

typedef __attribute__((ext_vector_type(16))) __bf16 v16bf;
typedef __attribute__((ext_vector_type(8)))  float  v8f;
typedef int v4i __attribute__((vector_size(16)));

#if __has_builtin(__builtin_amdgcn_global_load_async_to_lds_b128)
#define HAS_ASYNC_LDS 1
#else
#define HAS_ASYNC_LDS 0
#endif

union Frag16 {
    uint4 q[2];
    v16bf v;
};

__device__ __forceinline__ unsigned short f2bf(float f) {
    unsigned int u = __float_as_uint(f);
    unsigned int r = (u + 0x7FFFu + ((u >> 16) & 1u)) >> 16;  // RNE
    return (unsigned short)r;
}

__device__ __forceinline__ float gelu_tanh(float x) {
    const float c0 = 0.7978845608028654f;   // sqrt(2/pi)
    const float c1 = 0.044715f;
    return 0.5f * x * (1.0f + tanhf(c0 * (x + c1 * x * x * x)));
}

// ---------------- router path (tiny) ----------------

__global__ __launch_bounds__(256) void colmean_accum_kernel(
    const float* __restrict__ in, float* __restrict__ colmean) {
    int t = blockIdx.x * 256 + threadIdx.x;      // 0..B*H-1
    int b = t >> 10;                              // /H_
    int h = t & (H_ - 1);
    int s0 = blockIdx.y * (S_ / 16);
    const float* p = in + (size_t)b * S_ * H_ + (size_t)s0 * H_ + h;
    float acc = 0.f;
    for (int s = 0; s < S_ / 16; ++s) acc += p[(size_t)s * H_];
    atomicAdd(&colmean[t], acc * (1.0f / (float)S_));
}

__global__ __launch_bounds__(64) void router_kernel(
    const float* __restrict__ colmean, const float* __restrict__ w_router,
    const float* __restrict__ b_router, float* __restrict__ ew_out) {
    __shared__ float logits[B_][E_];
    int tid = threadIdx.x;
    if (tid < B_ * E_) {
        int b = tid >> 3, e = tid & 7;
        float acc = b_router[e];
        for (int h = 0; h < H_; ++h)
            acc += colmean[b * H_ + h] * w_router[h * E_ + e];
        logits[b][e] = acc;
    }
    __syncthreads();
    if (tid == 0) {
        float ew[E_];
        for (int e = 0; e < E_; ++e) ew[e] = 0.f;
        for (int b = 0; b < B_; ++b) {
            float m = logits[b][0];
            for (int e = 1; e < E_; ++e) m = fmaxf(m, logits[b][e]);
            float ex[E_]; float s = 0.f;
            for (int e = 0; e < E_; ++e) { ex[e] = __expf(logits[b][e] - m); s += ex[e]; }
            float inv = 1.0f / (s * (float)B_);
            for (int e = 0; e < E_; ++e) ew[e] += ex[e] * inv;
        }
        for (int e = 0; e < E_; ++e) ew_out[e] = ew[e];
    }
}

// ---------------- parameter merge (bandwidth-bound) ----------------

__global__ __launch_bounds__(256) void merge_bias_kernel(
    const float* __restrict__ b, const float* __restrict__ ew,
    float* __restrict__ out, int N) {
    int n = blockIdx.x * 256 + threadIdx.x;
    if (n >= N) return;
    float acc = 0.f;
    for (int e = 0; e < E_; ++e) acc += ew[e] * b[e * N + n];
    out[n] = acc;
}

// W: [E][K][N] f32 row-major.  WT: [N][K] bf16 (transposed merged weights).
__global__ __launch_bounds__(256) void merge_transpose_kernel(
    const float* __restrict__ W, const float* __restrict__ ew,
    unsigned short* __restrict__ WT, int K, int N) {
    __shared__ float tile[32][33];
    __shared__ float sew[E_];
    int tx = threadIdx.x, ty = threadIdx.y;
    int tid = ty * 32 + tx;
    if (tid < E_) sew[tid] = ew[tid];
    __syncthreads();
    int n0 = blockIdx.x * 32;
    int k0 = blockIdx.y * 32;
    for (int r = 0; r < 4; ++r) {
        int k = k0 + ty + 8 * r;
        int n = n0 + tx;
        const float* p = W + (size_t)k * N + n;
        float acc = 0.f;
        #pragma unroll
        for (int e = 0; e < E_; ++e) acc += sew[e] * p[(size_t)e * K * N];
        tile[ty + 8 * r][tx] = acc;
    }
    __syncthreads();
    for (int r = 0; r < 4; ++r) {
        int nl = ty + 8 * r;
        WT[(size_t)(n0 + nl) * K + k0 + tx] = f2bf(tile[tx][nl]);
    }
}

__global__ __launch_bounds__(256) void cvt_bf16_kernel(
    const float* __restrict__ in, unsigned short* __restrict__ out, int n4) {
    int i = blockIdx.x * 256 + threadIdx.x;
    if (i >= n4) return;
    float4 f = ((const float4*)in)[i];
    unsigned int lo = (unsigned int)f2bf(f.x) | ((unsigned int)f2bf(f.y) << 16);
    unsigned int hi = (unsigned int)f2bf(f.z) | ((unsigned int)f2bf(f.w) << 16);
    ((uint2*)out)[i] = make_uint2(lo, hi);
}

// ---------------- WMMA GEMM:  C[M][N] = act(A[M][K] @ BT[N][K]^T + bias) ----
// 256 threads = 8 wave32s. Block tile 128x128, wave tile 64x32, K-step 32,
// double-buffered LDS, async global->LDS staging when available.

template <bool GELU_BF16_OUT>
__global__ __launch_bounds__(256) void gemm_bf16_wmma_kernel(
    const unsigned short* __restrict__ A,    // [M][K] bf16
    const unsigned short* __restrict__ BT,   // [N][K] bf16
    const float* __restrict__ bias,          // [N]
    void* __restrict__ Cout, int M, int N, int K) {
    constexpr int LDT = 40;          // 32 halves + 8 pad
    constexpr int BUF = 128 * LDT;   // halves per buffer
    __shared__ __attribute__((aligned(16))) unsigned short sA[2 * BUF];
    __shared__ __attribute__((aligned(16))) unsigned short sB[2 * BUF];

    const int tid  = threadIdx.x;
    const int lane = tid & 31;
    const int wave = tid >> 5;
    const int wm = wave & 1;     // 0..1 -> 64-row strip
    const int wn = wave >> 1;    // 0..3 -> 32-col strip
    const int bm = blockIdx.y * 128;
    const int bn = blockIdx.x * 128;

    // staging: 512 x 16B chunks per buffer pair; thread covers chunk tid and tid+256
    const int srow = tid >> 2;               // 0..63
    const int scol = (tid & 3) << 3;         // 0,8,16,24 halves
    const unsigned short* gA0 = A  + (size_t)(bm + srow) * K + scol;
    const unsigned short* gA1 = A  + (size_t)(bm + srow + 64) * K + scol;
    const unsigned short* gB0 = BT + (size_t)(bn + srow) * K + scol;
    const unsigned short* gB1 = BT + (size_t)(bn + srow + 64) * K + scol;
    unsigned short* lA0 = &sA[srow * LDT + scol];
    unsigned short* lA1 = &sA[(srow + 64) * LDT + scol];
    unsigned short* lB0 = &sB[srow * LDT + scol];
    unsigned short* lB1 = &sB[(srow + 64) * LDT + scol];

    auto stage = [&](int buf, int kk) {
        const int lo = buf * BUF;
#if HAS_ASYNC_LDS
        __builtin_amdgcn_global_load_async_to_lds_b128(
            (v4i*)(gA0 + kk), (v4i*)(lA0 + lo), 0, 0);
        __builtin_amdgcn_global_load_async_to_lds_b128(
            (v4i*)(gA1 + kk), (v4i*)(lA1 + lo), 0, 0);
        __builtin_amdgcn_global_load_async_to_lds_b128(
            (v4i*)(gB0 + kk), (v4i*)(lB0 + lo), 0, 0);
        __builtin_amdgcn_global_load_async_to_lds_b128(
            (v4i*)(gB1 + kk), (v4i*)(lB1 + lo), 0, 0);
#else
        uint4 a0 = *(const uint4*)(gA0 + kk);
        uint4 a1 = *(const uint4*)(gA1 + kk);
        uint4 b0 = *(const uint4*)(gB0 + kk);
        uint4 b1 = *(const uint4*)(gB1 + kk);
        *(uint4*)(lA0 + lo) = a0;
        *(uint4*)(lA1 + lo) = a1;
        *(uint4*)(lB0 + lo) = b0;
        *(uint4*)(lB1 + lo) = b1;
#endif
    };

    v8f acc[4][2];
    #pragma unroll
    for (int i = 0; i < 4; ++i)
        #pragma unroll
        for (int j = 0; j < 2; ++j)
            acc[i][j] = (v8f){0.f, 0.f, 0.f, 0.f, 0.f, 0.f, 0.f, 0.f};

    const int nk = K >> 5;
    stage(0, 0);
    for (int kt = 0; kt < nk; ++kt) {
        const int cur = kt & 1;
#if HAS_ASYNC_LDS
        asm volatile("s_wait_asynccnt 0x0" ::: "memory");
#endif
        __syncthreads();
        if (kt + 1 < nk) stage(cur ^ 1, (kt + 1) << 5);

        const unsigned short* bufA = &sA[cur * BUF];
        const unsigned short* bufB = &sB[cur * BUF];

        // A fragment (16x32): lanes 0-15 row M, halves [0..7]+[16..23];
        //                     lanes 16-31 row M, halves [8..15]+[24..31]
        Frag16 af[4], bfr[2];
        #pragma unroll
        for (int mi = 0; mi < 4; ++mi) {
            int r = wm * 64 + mi * 16 + (lane & 15);
            const unsigned short* p = &bufA[r * LDT + ((lane >> 4) * 8)];
            af[mi].q[0] = *(const uint4*)p;
            af[mi].q[1] = *(const uint4*)(p + 16);
        }
        // B fragment (32x16, N-major in LDS): lanes 0-15 K[0..7]+[16..23], lanes 16-31 K[8..15]+[24..31]
        #pragma unroll
        for (int ni = 0; ni < 2; ++ni) {
            int n = wn * 32 + ni * 16 + (lane & 15);
            const unsigned short* p = &bufB[n * LDT + ((lane >> 4) * 8)];
            bfr[ni].q[0] = *(const uint4*)p;
            bfr[ni].q[1] = *(const uint4*)(p + 16);
        }
        #pragma unroll
        for (int mi = 0; mi < 4; ++mi)
            #pragma unroll
            for (int ni = 0; ni < 2; ++ni)
                acc[mi][ni] = __builtin_amdgcn_wmma_f32_16x16x32_bf16(
                    false, af[mi].v, false, bfr[ni].v,
                    (short)0, acc[mi][ni], false, false);
    }

    // epilogue: C/D layout -> lane holds col (lane&15), M = v + 8*(lane>>4)
    const int colb = bn + wn * 32 + (lane & 15);
    const int rowb = bm + wm * 64 + 8 * (lane >> 4);
    float bv0 = bias[colb];
    float bv1 = bias[colb + 16];
    #pragma unroll
    for (int mi = 0; mi < 4; ++mi) {
        #pragma unroll
        for (int ni = 0; ni < 2; ++ni) {
            int col = colb + ni * 16;
            float bv = ni ? bv1 : bv0;
            #pragma unroll
            for (int v = 0; v < 8; ++v) {
                int row = rowb + mi * 16 + v;
                float x = acc[mi][ni][v] + bv;
                if (GELU_BF16_OUT) {
                    ((unsigned short*)Cout)[(size_t)row * N + col] = f2bf(gelu_tanh(x));
                } else {
                    ((float*)Cout)[(size_t)row * N + col] = x;
                }
            }
        }
    }
}

// ---------------- launch ----------------

extern "C" void kernel_launch(void* const* d_in, const int* in_sizes, int n_in,
                              void* d_out, int out_size, void* d_ws, size_t ws_size,
                              hipStream_t stream) {
    const float* inputs   = (const float*)d_in[0];
    const float* w_router = (const float*)d_in[1];
    const float* b_router = (const float*)d_in[2];
    const float* W1 = (const float*)d_in[3];
    const float* b1 = (const float*)d_in[4];
    const float* W2 = (const float*)d_in[5];
    const float* b2 = (const float*)d_in[6];

    char* ws = (char*)d_ws;
    size_t off = 0;
    auto walloc = [&](size_t bytes) -> char* {
        char* p = ws + off;
        off = (off + bytes + 255) & ~(size_t)255;
        return p;
    };
    float* ew      = (float*)walloc(E_ * sizeof(float));
    float* colmean = (float*)walloc((size_t)B_ * H_ * sizeof(float));
    float* mb1     = (float*)walloc(FF_ * sizeof(float));
    float* mb2     = (float*)walloc(H_ * sizeof(float));
    unsigned short* Xbf = (unsigned short*)walloc((size_t)T_ * H_ * 2);
    unsigned short* W1T = (unsigned short*)walloc((size_t)FF_ * H_ * 2);
    unsigned short* W2T = (unsigned short*)walloc((size_t)H_ * FF_ * 2);
    unsigned short* Hbf = (unsigned short*)walloc((size_t)T_ * FF_ * 2);

    // router
    (void)hipMemsetAsync(colmean, 0, (size_t)B_ * H_ * sizeof(float), stream);
    colmean_accum_kernel<<<dim3(B_ * H_ / 256, 16), 256, 0, stream>>>(inputs, colmean);
    router_kernel<<<1, 64, 0, stream>>>(colmean, w_router, b_router, ew);

    // parameter merge (f32 read, bf16 transposed write)
    merge_bias_kernel<<<FF_ / 256, 256, 0, stream>>>(b1, ew, mb1, FF_);
    merge_bias_kernel<<<H_  / 256, 256, 0, stream>>>(b2, ew, mb2, H_);
    merge_transpose_kernel<<<dim3(FF_ / 32, H_ / 32), dim3(32, 8), 0, stream>>>(
        W1, ew, W1T, H_, FF_);
    merge_transpose_kernel<<<dim3(H_ / 32, FF_ / 32), dim3(32, 8), 0, stream>>>(
        W2, ew, W2T, FF_, H_);

    // activations to bf16
    cvt_bf16_kernel<<<(T_ * H_ / 4 + 255) / 256, 256, 0, stream>>>(inputs, Xbf, T_ * H_ / 4);

    // GEMM1 + GELU -> bf16 H, GEMM2 -> f32 out
    gemm_bf16_wmma_kernel<true ><<<dim3(FF_ / 128, T_ / 128), 256, 0, stream>>>(
        Xbf, W1T, mb1, Hbf, T_, FF_, H_);
    gemm_bf16_wmma_kernel<false><<<dim3(H_ / 128, T_ / 128), 256, 0, stream>>>(
        Hbf, W2T, mb2, d_out, T_, H_, FF_);
}
